// LSTMModel_18975165513887
// MI455X (gfx1250) — matrix operand order repaired and checked by reading, not compiled, not run on hardware
//
#include <hip/hip_runtime.h>
#include <hip/hip_bf16.h>

// ---------------------------------------------------------------------------
// LSTM LM forward for MI455X (gfx1250, wave32, WMMA).
// GEMMs via v_wmma_f32_16x16x32_bf16; big GEMM LDS-staged with fully-async
// global_load_async_to_lds_b128 double buffering (ASYNCcnt pipeline).
// ---------------------------------------------------------------------------

typedef __attribute__((ext_vector_type(16))) __bf16 v16bf;
typedef __attribute__((ext_vector_type(8)))  float  v8f;

struct __attribute__((aligned(16))) U128 { unsigned int x, y, z, w; };

union Frag32 {
    v16bf v;
    U128  q[2];
};

#define WMMA_BF16(A, B, C) \
    __builtin_amdgcn_wmma_f32_16x16x32_bf16(false, (A), false, (B), (short)0, (C), false, false)

// async 16B global -> LDS copy (per-lane), tracked by ASYNCcnt
#define ASYNC_LDS_B128(ldsaddr, gptr)                                   \
    asm volatile("global_load_async_to_lds_b128 %0, %1, off"            \
                 :: "v"(ldsaddr), "v"(gptr) : "memory")

// ---- constants --------------------------------------------------------------
#define TT     128
#define BB     16
#define HH     1024
#define GG     4096           // 4*H
#define VV     50257
#define VVPAD  50304          // 393 * 128
#define MM     (TT * BB)      // 2048

#define BMT    64             // GEMM tile M
#define BNT    128            // GEMM tile N
#define KST    32             // GEMM K step (WMMA K)

// ---- helpers ----------------------------------------------------------------
__device__ __forceinline__ unsigned short f2bf(float f) {
    unsigned int u = __float_as_uint(f);
    u += 0x7FFFu + ((u >> 16) & 1u);      // round-to-nearest-even
    return (unsigned short)(u >> 16);
}

__device__ __forceinline__ float sigmoidf_(float x) {
    return 1.0f / (1.0f + __expf(-x));
}

__device__ __forceinline__ unsigned lds_addr_of(const void* p) {
    // LDS aperture occupies the upper 32 bits of a generic pointer; the low
    // 32 bits are the wave-relative LDS byte address (ISA 10.2 aperture rules).
    return (unsigned)(unsigned long long)p;
}

// A fragment from global: 16x32 bf16, row-major A[M,K]
__device__ __forceinline__ v16bf load_a_frag(const unsigned short* __restrict__ A,
                                             int lda, int m0, int k0, int lane) {
    const unsigned short* p =
        A + (size_t)(m0 + (lane & 15)) * lda + k0 + ((lane >> 4) << 3);
    Frag32 f;
    f.q[0] = *(const U128*)(p);        // K = kb .. kb+7
    f.q[1] = *(const U128*)(p + 16);   // K = 16+kb .. 16+kb+7
    return f.v;
}

// B fragment from global: 32x16 bf16 where B[k][n] = W[n][k], W row-major [N,K]
__device__ __forceinline__ v16bf load_b_frag(const unsigned short* __restrict__ W,
                                             int ldb, int n0, int k0, int lane) {
    const unsigned short* p =
        W + (size_t)(n0 + (lane & 15)) * ldb + k0 + ((lane >> 4) << 3);
    Frag32 f;
    f.q[0] = *(const U128*)(p);
    f.q[1] = *(const U128*)(p + 16);
    return f.v;
}

// ---------------------------------------------------------------------------
// Staged GEMM: C[M,N] = A[M,K] @ W[N,K]^T + bias0 + bias1  (fp32 out)
// 256 thr = 8 waves; tile 64(M) x 128(N); wave = 32x32 (2x2 frags); K-step 32.
// Both tiles staged via global_load_async_to_lds_b128 (3 branch-free issues
// per thread per K-step). Double buffered, one split-barrier per K-step:
//   s_wait_asynccnt 0  ->  barrier  ->  issue next-tile asyncs  ->  WMMA.
// grid = (Npad/128, M/64); stores guarded by n < N; B rows must be padded.
// ---------------------------------------------------------------------------
__global__ void __launch_bounds__(256)
gemm_bf16_kernel(const unsigned short* __restrict__ A,
                 const unsigned short* __restrict__ W,
                 float* __restrict__ C,
                 int N, int K, int ldc,
                 const float* __restrict__ bias0,
                 const float* __restrict__ bias1) {
    __shared__ __align__(16) unsigned short Ash[2][BMT * KST];   // 2 x 4 KB
    __shared__ __align__(16) unsigned short Bsh[2][BNT * KST];   // 2 x 8 KB

    const int tid   = threadIdx.x;
    const int lane  = tid & 31;
    const int wave  = tid >> 5;
    const int m_blk = blockIdx.y * BMT;
    const int n_blk = blockIdx.x * BNT;
    const int wrow  = (wave >> 2) * 32;    // wave's row base inside tile
    const int wcol  = (wave & 3) * 32;     // wave's col base inside tile

    // staging map: 16B chunks. A: 256 chunks (1/thread); B: 512 chunks (2/thread)
    const int arow  = tid >> 2;
    const int aseg  = (tid & 3) << 3;      // 8-element segment offset
    const int brow0 = tid >> 2;            // B chunks tid and tid+256
    const int brow1 = 64 + (tid >> 2);
    const int bseg  = (tid & 3) << 3;

    auto stage = [&](int buf, int k0) {
        ASYNC_LDS_B128(lds_addr_of(&Bsh[buf][brow0 * KST + bseg]),
                       W + (size_t)(n_blk + brow0) * K + k0 + bseg);
        ASYNC_LDS_B128(lds_addr_of(&Bsh[buf][brow1 * KST + bseg]),
                       W + (size_t)(n_blk + brow1) * K + k0 + bseg);
        ASYNC_LDS_B128(lds_addr_of(&Ash[buf][arow * KST + aseg]),
                       A + (size_t)(m_blk + arow) * K + k0 + aseg);
    };

    v8f acc[2][2] = {{{0,0,0,0,0,0,0,0}, {0,0,0,0,0,0,0,0}},
                     {{0,0,0,0,0,0,0,0}, {0,0,0,0,0,0,0,0}}};

    const int nk = K / KST;
    stage(0, 0);

    const int kb8  = (lane >> 4) << 3;     // A frag K sub-base (0 / 8)
    const int kb16 = (lane >> 4) << 4;     // unused; B frags also use kb8 (0 / 8)
    const int ml   = lane & 15;

    for (int kt = 0; kt < nk; ++kt) {
        const int buf = kt & 1;
        asm volatile("s_wait_asynccnt 0x0" ::: "memory");  // own tile asyncs done
        __syncthreads();                                   // all waves staged

        if (kt + 1 < nk) stage((kt + 1) & 1, (kt + 1) * KST);

        Frag32 a0, a1, b0, b1;
        const int ar0 = (wrow + ml) * KST;
        const int ar1 = (wrow + 16 + ml) * KST;
        a0.q[0] = *(const U128*)&Ash[buf][ar0 + kb8];
        a0.q[1] = *(const U128*)&Ash[buf][ar0 + 16 + kb8];
        a1.q[0] = *(const U128*)&Ash[buf][ar1 + kb8];
        a1.q[1] = *(const U128*)&Ash[buf][ar1 + 16 + kb8];
        const int br0 = (wcol + ml) * KST;
        const int br1 = (wcol + 16 + ml) * KST;
        b0.q[0] = *(const U128*)&Bsh[buf][br0 + kb8];
        b0.q[1] = *(const U128*)&Bsh[buf][br0 + 16 + kb8];
        b1.q[0] = *(const U128*)&Bsh[buf][br1 + kb8];
        b1.q[1] = *(const U128*)&Bsh[buf][br1 + 16 + kb8];

        acc[0][0] = WMMA_BF16(a0.v, b0.v, acc[0][0]);
        acc[0][1] = WMMA_BF16(a0.v, b1.v, acc[0][1]);
        acc[1][0] = WMMA_BF16(a1.v, b0.v, acc[1][0]);
        acc[1][1] = WMMA_BF16(a1.v, b1.v, acc[1][1]);
    }

    const int rsel = (lane >> 4) * 8;
    for (int j = 0; j < 2; ++j) {
        int n = n_blk + wcol + j * 16 + ml;
        if (n >= N) continue;
        float bv = 0.0f;
        if (bias0) bv += bias0[n];
        if (bias1) bv += bias1[n];
        for (int i = 0; i < 2; ++i) {
            int mbase = m_blk + wrow + i * 16 + rsel;
            #pragma unroll
            for (int r = 0; r < 8; ++r)
                C[(size_t)(mbase + r) * ldc + n] = acc[i][j][r] + bv;
        }
    }
}

// ---------------------------------------------------------------------------
// Recurrent step GEMM: gates[16,4096] += Hbf[16,1024] @ Whh[4096,1024]^T
// M = 16 -> single m-tile; direct-global (latency-bound, Whh L2-resident).
// ---------------------------------------------------------------------------
__global__ void __launch_bounds__(256)
lstm_gates_gemm(const unsigned short* __restrict__ Hbf,
                const unsigned short* __restrict__ Whh,
                float* __restrict__ gates) {
    const int K = HH;
    const int lane = threadIdx.x & 31;
    const int wave = threadIdx.x >> 5;
    const int n_wave = blockIdx.x * 256 + wave * 32;

    v8f acc[2] = {{0,0,0,0,0,0,0,0}, {0,0,0,0,0,0,0,0}};

    for (int k0 = 0; k0 < K; k0 += 32) {
        if (k0 + 64 < K) {   // near-scope prefetch of upcoming Whh rows
            __builtin_prefetch(Whh + (size_t)(n_wave + (lane & 15)) * K + k0 + 64, 0, 3);
            __builtin_prefetch(Whh + (size_t)(n_wave + 16 + (lane & 15)) * K + k0 + 64, 0, 3);
        }
        v16bf a  = load_a_frag(Hbf, K, 0, k0, lane);
        v16bf b0 = load_b_frag(Whh, K, n_wave,      k0, lane);
        v16bf b1 = load_b_frag(Whh, K, n_wave + 16, k0, lane);
        acc[0] = WMMA_BF16(a, b0, acc[0]);
        acc[1] = WMMA_BF16(a, b1, acc[1]);
    }

    const int col = lane & 15;
    const int rsel = (lane >> 4) * 8;
    for (int j = 0; j < 2; ++j) {
        int n = n_wave + j * 16 + col;
        for (int r = 0; r < 8; ++r) {
            float* p = gates + (size_t)(rsel + r) * GG + n;
            *p = *p + acc[j][r];
        }
    }
}

// ---------------------------------------------------------------------------
// LSTM pointwise cell: i,f,g,o -> c_new, h_new   (16*1024 threads)
// ---------------------------------------------------------------------------
__global__ void lstm_cell_kernel(const float* __restrict__ gates,
                                 float* __restrict__ c,
                                 float* __restrict__ h,
                                 unsigned short* __restrict__ hbf,
                                 unsigned short* __restrict__ ybf_t) {
    int idx = blockIdx.x * blockDim.x + threadIdx.x;
    if (idx >= BB * HH) return;
    int b = idx >> 10, j = idx & (HH - 1);
    const float* g = gates + (size_t)b * GG;
    float ig = sigmoidf_(g[j]);
    float fg = sigmoidf_(g[HH + j]);
    float gg = tanhf(g[2 * HH + j]);
    float og = sigmoidf_(g[3 * HH + j]);
    float cn = fg * c[idx] + ig * gg;
    float hn = og * tanhf(cn);
    c[idx] = cn;
    h[idx] = hn;
    unsigned short hb = f2bf(hn);
    hbf[idx] = hb;
    ybf_t[idx] = hb;
}

// ---------------------------------------------------------------------------
// Misc small kernels
// ---------------------------------------------------------------------------
__global__ void cvt_f32_bf16(const float* __restrict__ src,
                             unsigned short* __restrict__ dst,
                             long long nsrc, long long ntot) {
    long long i = (long long)blockIdx.x * blockDim.x + threadIdx.x;
    long long stride = (long long)gridDim.x * blockDim.x;
    for (; i < ntot; i += stride)
        dst[i] = (i < nsrc) ? f2bf(src[i]) : (unsigned short)0;
}

__global__ void embed_kernel(const int* __restrict__ tokens,
                             const float* __restrict__ embW,
                             unsigned short* __restrict__ xbf) {
    int row = blockIdx.x;                       // t*B + b, 0..2047
    int tok = tokens[row];
    const float* src = embW + (size_t)tok * HH;
    unsigned short* dst = xbf + (size_t)row * HH;
    for (int i = threadIdx.x; i < HH; i += blockDim.x)
        dst[i] = f2bf(src[i]);
}

__global__ void zero_u32(unsigned int* __restrict__ p, int n) {
    int i = blockIdx.x * blockDim.x + threadIdx.x;
    int stride = gridDim.x * blockDim.x;
    for (; i < n; i += stride) p[i] = 0u;
}

__global__ void copy_hc(const float* __restrict__ h, const float* __restrict__ c,
                        float* __restrict__ oh, float* __restrict__ oc) {
    int i = blockIdx.x * blockDim.x + threadIdx.x;
    if (i < BB * HH) { oh[i] = h[i]; oc[i] = c[i]; }
}

// ---------------------------------------------------------------------------
// In-place log-softmax per row (online max/sum, LDS tree reduction)
// ---------------------------------------------------------------------------
__global__ void __launch_bounds__(256)
log_softmax_rows(float* __restrict__ logits, int V) {
    float* p = logits + (size_t)blockIdx.x * V;
    float m = -3.0e38f, s = 0.0f;
    for (int i = threadIdx.x; i < V; i += blockDim.x) {
        float x = p[i];
        float nm = fmaxf(m, x);
        s = s * __expf(m - nm) + __expf(x - nm);
        m = nm;
    }
    __shared__ float sm[256];
    __shared__ float ss[256];
    sm[threadIdx.x] = m; ss[threadIdx.x] = s;
    __syncthreads();
    for (int off = 128; off > 0; off >>= 1) {
        if ((int)threadIdx.x < off) {
            float m2 = sm[threadIdx.x + off], s2 = ss[threadIdx.x + off];
            float M2 = fmaxf(sm[threadIdx.x], m2);
            ss[threadIdx.x] = ss[threadIdx.x] * __expf(sm[threadIdx.x] - M2)
                            + s2 * __expf(m2 - M2);
            sm[threadIdx.x] = M2;
        }
        __syncthreads();
    }
    float L = sm[0] + __logf(ss[0]);
    for (int i = threadIdx.x; i < V; i += blockDim.x)
        p[i] -= L;
}

// ---------------------------------------------------------------------------
// Host-side orchestration
// ---------------------------------------------------------------------------
extern "C" void kernel_launch(void* const* d_in, const int* in_sizes, int n_in,
                              void* d_out, int out_size, void* d_ws, size_t ws_size,
                              hipStream_t stream) {
    const int*   tokens = (const int*)  d_in[0];
    const float* embW   = (const float*)d_in[1];
    const float* Wih    = (const float*)d_in[2];   // [2,4096,1024]
    const float* Whh    = (const float*)d_in[3];   // [2,4096,1024]
    const float* bih    = (const float*)d_in[4];   // [2,4096]
    const float* bhh    = (const float*)d_in[5];   // [2,4096]
    const float* decW   = (const float*)d_in[6];   // [50257,1024]
    const float* decb   = (const float*)d_in[7];   // [50257]
    float* out = (float*)d_out;

    // ---- workspace carving (256B aligned) ----
    char* base = (char*)d_ws;
    size_t off = 0;
    auto carve = [&](size_t bytes) -> char* {
        char* p = base + off;
        off = (off + bytes + 255) & ~(size_t)255;
        return p;
    };
    const size_t WSZ = (size_t)2 * GG * HH;                    // per-weight elems
    unsigned short* wih_bf = (unsigned short*)carve(WSZ * 2);
    unsigned short* whh_bf = (unsigned short*)carve(WSZ * 2);
    unsigned short* dec_bf = (unsigned short*)carve((size_t)VVPAD * HH * 2);
    unsigned short* xbufA  = (unsigned short*)carve((size_t)MM * HH * 2);
    unsigned short* xbufB  = (unsigned short*)carve((size_t)MM * HH * 2);
    float*          xw     = (float*)         carve((size_t)MM * GG * 4);
    float*          h_f32  = (float*)         carve((size_t)BB * HH * 4);
    float*          c_f32  = (float*)         carve((size_t)BB * HH * 4);
    unsigned short* h_bf   = (unsigned short*)carve((size_t)BB * HH * 2);

    // ---- 1) weight conversion fp32 -> bf16 (dec padded with zero rows) ----
    cvt_f32_bf16<<<4096, 256, 0, stream>>>(Wih, wih_bf, (long long)WSZ, (long long)WSZ);
    cvt_f32_bf16<<<4096, 256, 0, stream>>>(Whh, whh_bf, (long long)WSZ, (long long)WSZ);
    cvt_f32_bf16<<<8192, 256, 0, stream>>>(decW, dec_bf,
                                           (long long)VV * HH, (long long)VVPAD * HH);

    // ---- 2) embedding gather -> bf16 activations ----
    embed_kernel<<<MM, 256, 0, stream>>>(tokens, embW, xbufA);

    const size_t OUTBASE = (size_t)MM * VV;   // start of hidden-state outputs
    unsigned short* xin  = xbufA;
    unsigned short* xout = xbufB;

    for (int l = 0; l < 2; ++l) {
        // xw = x @ Wih^T + bih + bhh   (staged WMMA GEMM, M=2048 N=4096 K=1024)
        gemm_bf16_kernel<<<dim3(GG / BNT, MM / BMT), 256, 0, stream>>>(
            xin, wih_bf + (size_t)l * WSZ, xw,
            GG, HH, GG, bih + (size_t)l * GG, bhh + (size_t)l * GG);

        // zero h_f32, c_f32, h_bf (contiguous carve region)
        zero_u32<<<160, 256, 0, stream>>>((unsigned int*)h_f32,
                                          (BB * HH * 4 + BB * HH * 4 + BB * HH * 2) / 4);

        // sequential scan over T
        for (int t = 0; t < TT; ++t) {
            float* gates_t = xw + (size_t)t * BB * GG;
            lstm_gates_gemm<<<GG / 256, 256, 0, stream>>>(
                h_bf, whh_bf + (size_t)l * WSZ, gates_t);
            lstm_cell_kernel<<<(BB * HH) / 256, 256, 0, stream>>>(
                gates_t, c_f32, h_f32, h_bf, xout + (size_t)t * BB * HH);
        }

        // export hT / cT:  out layout = [logits][h(2,16,1024)][c(2,16,1024)]
        copy_hc<<<(BB * HH) / 256, 256, 0, stream>>>(
            h_f32, c_f32,
            out + OUTBASE + (size_t)l * BB * HH,
            out + OUTBASE + (size_t)2 * BB * HH + (size_t)l * BB * HH);

        unsigned short* tmp = xin; xin = xout; xout = tmp;
    }

    // ---- 3) decoder GEMM straight into d_out: [2048,50257] ----
    gemm_bf16_kernel<<<dim3(VVPAD / BNT, MM / BMT), 256, 0, stream>>>(
        xin, dec_bf, out, VV, HH, VV, decb, nullptr);

    // ---- 4) in-place log-softmax ----
    log_softmax_rows<<<MM, 256, 0, stream>>>(out, VV);
}
